// FeedForward_37349035606276
// MI455X (gfx1250) — matrix operand order repaired and checked by reading, not compiled, not run on hardware
//
#include <hip/hip_runtime.h>
#include <hip/hip_bf16.h>

// ---------------- problem constants ----------------
constexpr int Dm   = 2048;   // model dim
constexpr int DFF  = 5632;   // ffn dim
constexpr int E    = 8;      // experts
constexpr int R    = 16;     // lora rank
constexpr int NMOE = 1024;   // tokens in MoE half
constexpr int NTOK = 2048;   // total token rows in data
constexpr int HROW = 3072;   // slot0 (1024) + slot1 (1024) + dense (1024)
constexpr float SCL = 2.0f;  // 32/16

// GEMM tiling
constexpr int BM = 128;
constexpr int BN = 128;
constexpr int BK = 32;

typedef unsigned short u16;
typedef __attribute__((ext_vector_type(16))) __bf16 v16bf;
typedef __attribute__((ext_vector_type(8)))  float  v8f;
typedef __attribute__((ext_vector_type(4)))  unsigned int u32x4;
typedef __attribute__((ext_vector_type(8)))  int         i32x8;
typedef __attribute__((ext_vector_type(4)))  int         i32x4;

union FragU { v16bf v; uint4 q[2]; };

// fp32 -> bf16 (round to nearest even)
__device__ __forceinline__ u16 f2bf(float f) {
    unsigned int u = __float_as_uint(f);
    unsigned int r = u + 0x7FFFu + ((u >> 16) & 1u);
    return (u16)(r >> 16);
}
__device__ __forceinline__ float bf2f(u16 b) {
    return __uint_as_float(((unsigned int)b) << 16);
}

// ---- Tensor Data Mover: 2D tile (tile_rows x BK bf16) global -> LDS ----
// D# per CDNA5 ISA. Group0: count=1, lds_addr,
// global_addr[56:0], type=2. Group1: data_size=1 (2B), tensor_dim0=ld,
// tensor_dim1=tile_rows, tile_dim0=BK, tile_dim1=tile_rows, dim0_stride=ld.
// Groups 2/3 zero (<=2D tensor). Tracked by TENSORcnt.
// Toolchain uses the 6-arg builtin: (g0, g1, g2, g3, g4, cpol).
__device__ __forceinline__ void tdm_load_tile(unsigned lds_addr,
                                              const u16* gbase, long elem_off,
                                              unsigned ld_elems, unsigned tile_rows) {
    unsigned long long ga =
        (unsigned long long)(size_t)gbase + 2ull * (unsigned long long)elem_off;
    u32x4 g0;
    g0[0] = 1u;                                        // count=1, user D#
    g0[1] = lds_addr;                                  // LDS byte address
    g0[2] = (unsigned)(ga & 0xFFFFFFFFu);              // global_addr[31:0]
    g0[3] = (unsigned)((ga >> 32) & 0x1FFFFFFu)        // global_addr[56:32]
            | 0x80000000u;                             // type=2 ("image")
    i32x8 g1;
    g1[0] = 0x10000;                                   // data_size=1 (2 bytes)
    g1[1] = (int)((ld_elems & 0xFFFFu) << 16);         // tensor_dim0[15:0]
    g1[2] = (int)((ld_elems >> 16) |
                  ((tile_rows & 0xFFFFu) << 16));      // dim0[31:16] | dim1[15:0]
    g1[3] = (int)((tile_rows >> 16) |
                  ((unsigned)BK << 16));               // dim1[31:16] | tile_dim0
    g1[4] = (int)tile_rows;                            // tile_dim1 (tile_dim2=0)
    g1[5] = (int)ld_elems;                             // tensor_dim0_stride[31:0]
    g1[6] = 0;                                         // stride[47:32] | dim1_stride lo
    g1[7] = 0;
    i32x4 z4 = {0, 0, 0, 0};
    i32x8 z8 = {0, 0, 0, 0, 0, 0, 0, 0};
    __builtin_amdgcn_tensor_load_to_lds(g0, g1, z4, z4, z8, 0);
}

__device__ __forceinline__ unsigned lds_byte_addr(const void* p) {
    // generic address of LDS: addr[31:0] is the LDS byte address (ISA §10.2)
    return (unsigned)(size_t)p;
}

// A-matrix 16x32 bf16 fragment from an LDS tile (row-major, ld = BK):
// lane half=l>>4: elems 0..7 <- K = 8*half+(0..7), elems 8..15 <- K = 16+8*half+(0..7)
__device__ __forceinline__ v16bf lds_a_frag(const u16* __restrict__ s, int row, int lane) {
    const int half = lane >> 4;
    const u16* rp = s + row * BK + half * 8;
    FragU f;
    f.q[0] = *reinterpret_cast<const uint4*>(rp);
    f.q[1] = *reinterpret_cast<const uint4*>(rp + 16);
    return f.v;
}

// B-matrix 32x16 bf16 fragment from an LDS tile (row-major (N,K), ld = BK):
// lane half=l>>4: elems 0..7 <- K = 8*half+(0..7), elems 8..15 <- K = 16+8*half+(0..7)
__device__ __forceinline__ v16bf lds_b_frag(const u16* __restrict__ s, int col, int lane) {
    const int half = lane >> 4;
    const u16* rp = s + col * BK + half * 8;
    FragU f;
    f.q[0] = *reinterpret_cast<const uint4*>(rp);
    f.q[1] = *reinterpret_cast<const uint4*>(rp + 16);
    return f.v;
}

__device__ __forceinline__ v8f wmma_bf16(v16bf a, v16bf b, v8f c) {
    return __builtin_amdgcn_wmma_f32_16x16x32_bf16(false, a, false, b, (short)0, c,
                                                   false, false);
}

__device__ __forceinline__ float silu(float x) {
    return x / (1.0f + __expf(-x));
}

__device__ __forceinline__ float wave_sum(float s) {
    #pragma unroll
    for (int off = 16; off > 0; off >>= 1) s += __shfl_xor(s, off, 32);
    return s;
}

// ---------------- kernels ----------------

// fp32 -> bf16 bulk convert (grid-stride)
__global__ void moe_cvt_bf16(const float* __restrict__ src, u16* __restrict__ dst, long n) {
    long i = (long)blockIdx.x * blockDim.x + threadIdx.x;
    long stride = (long)gridDim.x * blockDim.x;
    for (; i < n; i += stride) dst[i] = f2bf(src[i]);
}

// logits[n,e] = x[n,:] . gate_W[e,:]   (one wave per (token,expert))
__global__ __launch_bounds__(256) void moe_gate(const float* __restrict__ data,
                                                const float* __restrict__ gateW,
                                                float* __restrict__ logits) {
    const int n = blockIdx.x;
    const int e = threadIdx.x >> 5;
    const int lane = threadIdx.x & 31;
    const float* xr = data + (long)n * Dm;
    const float* wr = gateW + (long)e * Dm;
    float s = 0.f;
    for (int c = lane; c < Dm; c += 32) s += xr[c] * wr[c];
    s = wave_sum(s);
    if (lane == 0) logits[n * E + e] = s;
}

// top-2 of softmax + renormalized weights (softmax monotone -> top-2 of logits)
__global__ void moe_topk(const float* __restrict__ logits,
                         int* __restrict__ eidx, float* __restrict__ ew) {
    int n = blockIdx.x * blockDim.x + threadIdx.x;
    if (n >= NMOE) return;
    float l[E];
    #pragma unroll
    for (int e = 0; e < E; ++e) l[e] = logits[n * E + e];
    int i0 = 0;
    #pragma unroll
    for (int e = 1; e < E; ++e) if (l[e] > l[i0]) i0 = e;
    int i1 = (i0 == 0) ? 1 : 0;
    #pragma unroll
    for (int e = 0; e < E; ++e) { if (e == i0) continue; if (l[e] > l[i1]) i1 = e; }
    float p1 = __expf(l[i1] - l[i0]);     // p0 = 1
    float inv = 1.0f / (1.0f + p1);
    eidx[n * 2 + 0] = i0;  eidx[n * 2 + 1] = i1;
    ew[n * 2 + 0] = inv;   ew[n * 2 + 1] = p1 * inv;
}

// T1[hr,r] = x_row(hr) . Arow ; T3 likewise. hr: 0..1023 slot0, 1024..2047 slot1,
// 2048..3071 dense (uses la1/la3). 8 waves/block, 32 (r,matrix) tasks.
__global__ __launch_bounds__(256) void moe_lora_x(const float* __restrict__ data,
                                                  const float* __restrict__ A1,
                                                  const float* __restrict__ A3,
                                                  const float* __restrict__ la1,
                                                  const float* __restrict__ la3,
                                                  const int* __restrict__ eidx,
                                                  float* __restrict__ T1,
                                                  float* __restrict__ T3) {
    const int hr = blockIdx.x;
    const int wave = threadIdx.x >> 5;
    const int lane = threadIdx.x & 31;
    const float* xrow = data + (long)((hr < 2048) ? (hr & 1023) : (hr - 1024)) * Dm;
    for (int task = wave; task < 32; task += 8) {
        const int r = task & 15;
        const int which = task >> 4;   // 0: A1->T1, 1: A3->T3
        const float* arow;
        if (hr < 2048) {
            const int tr = hr & 1023, slot = hr >> 10;
            const int e = eidx[tr * 2 + slot];
            arow = (which ? A3 : A1) + ((long)e * R + r) * Dm;
        } else {
            arow = (which ? la3 : la1) + (long)r * Dm;
        }
        float s = 0.f;
        for (int c = lane; c < Dm; c += 32) s += xrow[c] * arow[c];
        s = wave_sum(s);
        if (lane == 0) (which ? T3 : T1)[hr * R + r] = s;
    }
}

// Fused up/gate tiled GEMM (+ per-expert LoRA delta + SiLU) -> Hbf (bf16 3072x5632)
// Workgroup: 256 threads (8 waves), tile BM x BN, LDS double-buffered, K-step 32.
// Staging: wave 0 issues TDM tensor_load_to_lds; compute waves consume LDS frags.
__global__ __launch_bounds__(256) void moe_gemm_gu(const u16* __restrict__ Xbf,
                                                   const u16* __restrict__ W1bf,
                                                   const u16* __restrict__ W3bf,
                                                   const float* __restrict__ T1,
                                                   const float* __restrict__ T3,
                                                   const float* __restrict__ B1,
                                                   const float* __restrict__ B3,
                                                   const float* __restrict__ lb1,
                                                   const float* __restrict__ lb3,
                                                   const int* __restrict__ eidx,
                                                   u16* __restrict__ Hbf) {
    __shared__ __align__(16) u16 sA [2][BM * BK];
    __shared__ __align__(16) u16 sBg[2][BN * BK];
    __shared__ __align__(16) u16 sBu[2][BN * BK];

    const int ftile = blockIdx.x;            // N block (DFF/BN)
    const int ttile = blockIdx.y;            // M block (NTOK/BM)
    const int t = threadIdx.x;
    const int lane = t & 31;
    const int w = t >> 5;
    const int wm = w & 3;                    // M subtile pair: rows wm*32..+32
    const int wn = w >> 2;                   // N subtile quad: cols wn*64..+64
    const int nloc = lane & 15;
    const int half = lane >> 4;

    const int row0 = ttile * BM;             // token rows
    const int col0 = ftile * BN;             // f columns

    v8f accg[2][4] = {};
    v8f accu[2][4] = {};

    // prologue: TDM-stage kb = 0 into buffer 0
    if (w == 0) {
        tdm_load_tile(lds_byte_addr(&sA [0][0]), Xbf,  (long)row0 * Dm, Dm, BM);
        tdm_load_tile(lds_byte_addr(&sBg[0][0]), W1bf, (long)col0 * Dm, Dm, BN);
        tdm_load_tile(lds_byte_addr(&sBu[0][0]), W3bf, (long)col0 * Dm, Dm, BN);
        __builtin_amdgcn_s_wait_tensorcnt(0);
    }
    __syncthreads();

    for (int kb = 0; kb < Dm; kb += BK) {
        const int cur = (kb / BK) & 1;
        const int nxt = cur ^ 1;
        const bool hasNext = (kb + BK) < Dm;

        if (w == 0 && hasNext) {             // async prefetch of next K-slab
            tdm_load_tile(lds_byte_addr(&sA [nxt][0]), Xbf,
                          (long)row0 * Dm + (kb + BK), Dm, BM);
            tdm_load_tile(lds_byte_addr(&sBg[nxt][0]), W1bf,
                          (long)col0 * Dm + (kb + BK), Dm, BN);
            tdm_load_tile(lds_byte_addr(&sBu[nxt][0]), W3bf,
                          (long)col0 * Dm + (kb + BK), Dm, BN);
        }

        v16bf afr[2];
        #pragma unroll
        for (int mi = 0; mi < 2; ++mi)
            afr[mi] = lds_a_frag(sA[cur], wm * 32 + mi * 16 + nloc, lane);
        #pragma unroll
        for (int ni = 0; ni < 4; ++ni) {
            v16bf bg = lds_b_frag(sBg[cur], wn * 64 + ni * 16 + nloc, lane);
            v16bf bu = lds_b_frag(sBu[cur], wn * 64 + ni * 16 + nloc, lane);
            #pragma unroll
            for (int mi = 0; mi < 2; ++mi) {
                accg[mi][ni] = wmma_bf16(afr[mi], bg, accg[mi][ni]);
                accu[mi][ni] = wmma_bf16(afr[mi], bu, accu[mi][ni]);
            }
        }

        if (w == 0 && hasNext) __builtin_amdgcn_s_wait_tensorcnt(0);
        __syncthreads();
    }

    // epilogue: per-element LoRA delta + SiLU, write bf16 H rows
    #pragma unroll
    for (int mi = 0; mi < 2; ++mi) {
        #pragma unroll
        for (int ni = 0; ni < 4; ++ni) {
            #pragma unroll
            for (int v = 0; v < 8; ++v) {
                const int tr = row0 + wm * 32 + mi * 16 + v + 8 * half;
                const int f  = col0 + wn * 64 + ni * 16 + nloc;
                const float g = accg[mi][ni][v];
                const float u = accu[mi][ni][v];
                if (tr < NMOE) {                       // MoE: two expert slots
                    #pragma unroll
                    for (int slot = 0; slot < 2; ++slot) {
                        const int hr = slot * NMOE + tr;
                        const int e = eidx[tr * 2 + slot];
                        const float* t1 = T1 + hr * R;
                        const float* t3 = T3 + hr * R;
                        const float* b1 = B1 + ((long)e * DFF + f) * R;
                        const float* b3 = B3 + ((long)e * DFF + f) * R;
                        float dg = 0.f, du = 0.f;
                        #pragma unroll
                        for (int r = 0; r < R; ++r) { dg += t1[r] * b1[r]; du += t3[r] * b3[r]; }
                        const float h = silu(g + SCL * dg) * (u + SCL * du);
                        Hbf[(long)hr * DFF + f] = f2bf(h);
                    }
                } else {                               // dense half
                    const int hr = tr + 1024;          // 2048..3071
                    const float* t1 = T1 + hr * R;
                    const float* t3 = T3 + hr * R;
                    const float* b1 = lb1 + (long)f * R;
                    const float* b3 = lb3 + (long)f * R;
                    float dg = 0.f, du = 0.f;
                    #pragma unroll
                    for (int r = 0; r < R; ++r) { dg += t1[r] * b1[r]; du += t3[r] * b3[r]; }
                    const float h = silu(g + SCL * dg) * (u + SCL * du);
                    Hbf[(long)hr * DFF + f] = f2bf(h);
                }
            }
        }
    }
}

// T2[hr,r] = H[hr,:] . A2row (rank-16 contraction over DFF)
__global__ __launch_bounds__(256) void moe_t2(const u16* __restrict__ Hbf,
                                              const float* __restrict__ A2,
                                              const float* __restrict__ la2,
                                              const int* __restrict__ eidx,
                                              float* __restrict__ T2) {
    const int hr = blockIdx.x;
    const int wave = threadIdx.x >> 5;
    const int lane = threadIdx.x & 31;
    const u16* hrow = Hbf + (long)hr * DFF;
    for (int r = wave; r < R; r += 8) {
        const float* arow;
        if (hr < 2048) {
            const int tr = hr & 1023, slot = hr >> 10;
            const int e = eidx[tr * 2 + slot];
            arow = A2 + ((long)e * R + r) * DFF;
        } else {
            arow = la2 + (long)r * DFF;
        }
        float s = 0.f;
        for (int c = lane; c < DFF; c += 32) s += bf2f(hrow[c]) * arow[c];
        s = wave_sum(s);
        if (lane == 0) T2[hr * R + r] = s;
    }
}

// Down tiled GEMM: Y[hr,d] = H[hr,:] . W2[d,:] + SCL * (T2[hr,:] . B2sel[d,:])
__global__ __launch_bounds__(256) void moe_gemm_y(const u16* __restrict__ Hbf,
                                                  const u16* __restrict__ W2bf,
                                                  const float* __restrict__ T2,
                                                  const float* __restrict__ B2,
                                                  const float* __restrict__ lb2,
                                                  const int* __restrict__ eidx,
                                                  float* __restrict__ Yadj) {
    __shared__ __align__(16) u16 sA[2][BM * BK];
    __shared__ __align__(16) u16 sB[2][BN * BK];

    const int dtile = blockIdx.x;            // N block (Dm/BN)
    const int htile = blockIdx.y;            // M block (HROW/BM)
    const int t = threadIdx.x;
    const int lane = t & 31;
    const int w = t >> 5;
    const int wm = w & 3;
    const int wn = w >> 2;
    const int nloc = lane & 15;
    const int half = lane >> 4;

    const int row0 = htile * BM;             // H rows
    const int col0 = dtile * BN;             // output d columns

    v8f acc[2][4] = {};

    if (w == 0) {
        tdm_load_tile(lds_byte_addr(&sA[0][0]), Hbf,  (long)row0 * DFF, DFF, BM);
        tdm_load_tile(lds_byte_addr(&sB[0][0]), W2bf, (long)col0 * DFF, DFF, BN);
        __builtin_amdgcn_s_wait_tensorcnt(0);
    }
    __syncthreads();

    for (int kb = 0; kb < DFF; kb += BK) {
        const int cur = (kb / BK) & 1;
        const int nxt = cur ^ 1;
        const bool hasNext = (kb + BK) < DFF;

        if (w == 0 && hasNext) {
            tdm_load_tile(lds_byte_addr(&sA[nxt][0]), Hbf,
                          (long)row0 * DFF + (kb + BK), DFF, BM);
            tdm_load_tile(lds_byte_addr(&sB[nxt][0]), W2bf,
                          (long)col0 * DFF + (kb + BK), DFF, BN);
        }

        v16bf afr[2];
        #pragma unroll
        for (int mi = 0; mi < 2; ++mi)
            afr[mi] = lds_a_frag(sA[cur], wm * 32 + mi * 16 + nloc, lane);
        #pragma unroll
        for (int ni = 0; ni < 4; ++ni) {
            v16bf bf = lds_b_frag(sB[cur], wn * 64 + ni * 16 + nloc, lane);
            #pragma unroll
            for (int mi = 0; mi < 2; ++mi)
                acc[mi][ni] = wmma_bf16(afr[mi], bf, acc[mi][ni]);
        }

        if (w == 0 && hasNext) __builtin_amdgcn_s_wait_tensorcnt(0);
        __syncthreads();
    }

    #pragma unroll
    for (int mi = 0; mi < 2; ++mi) {
        #pragma unroll
        for (int ni = 0; ni < 4; ++ni) {
            #pragma unroll
            for (int v = 0; v < 8; ++v) {
                const int hr = row0 + wm * 32 + mi * 16 + v + 8 * half;
                const int d  = col0 + wn * 64 + ni * 16 + nloc;
                const float* t2 = T2 + hr * R;
                const float* b2;
                if (hr < 2048) {
                    const int tr = hr & 1023, slot = hr >> 10;
                    const int e = eidx[tr * 2 + slot];
                    b2 = B2 + ((long)e * Dm + d) * R;
                } else {
                    b2 = lb2 + (long)d * R;
                }
                float del = 0.f;
                #pragma unroll
                for (int r = 0; r < R; ++r) del += t2[r] * b2[r];
                Yadj[(long)hr * Dm + d] = acc[mi][ni][v] + SCL * del;
            }
        }
    }
}

// Final: MoE rows = w0*Y[slot0] + w1*Y[slot1]; dense rows copied.
__global__ void moe_combine(const float* __restrict__ Yadj,
                            const float* __restrict__ ew,
                            float* __restrict__ out) {
    long idx = (long)blockIdx.x * blockDim.x + threadIdx.x;
    if (idx >= (long)NTOK * Dm) return;
    const long n = idx / Dm;
    const int d = (int)(idx - n * Dm);
    if (n < NMOE) {
        const float w0 = ew[n * 2 + 0], w1 = ew[n * 2 + 1];
        out[idx] = w0 * Yadj[n * Dm + d] + w1 * Yadj[(n + NMOE) * Dm + d];
    } else {
        out[idx] = Yadj[(n + NMOE) * Dm + d];   // dense rows live at 2048..3071
    }
}

// ---------------- launch ----------------
extern "C" void kernel_launch(void* const* d_in, const int* in_sizes, int n_in,
                              void* d_out, int out_size, void* d_ws, size_t ws_size,
                              hipStream_t stream) {
    const float* data  = (const float*)d_in[0];
    const float* W1    = (const float*)d_in[1];
    const float* W3    = (const float*)d_in[2];
    const float* W2    = (const float*)d_in[3];
    const float* gateW = (const float*)d_in[4];
    const float* A1    = (const float*)d_in[5];
    const float* B1    = (const float*)d_in[6];
    const float* A3    = (const float*)d_in[7];
    const float* B3    = (const float*)d_in[8];
    const float* A2    = (const float*)d_in[9];
    const float* B2    = (const float*)d_in[10];
    const float* la1   = (const float*)d_in[11];
    const float* lb1   = (const float*)d_in[12];
    const float* la3   = (const float*)d_in[13];
    const float* lb3   = (const float*)d_in[14];
    const float* la2   = (const float*)d_in[15];
    const float* lb2   = (const float*)d_in[16];

    float* out    = (float*)d_out;
    float* logits = out + (size_t)4 * 512 * Dm;   // tuple tail

    // workspace layout (~138 MB)
    char* ws = (char*)d_ws;
    size_t off = 0;
    auto alloc = [&](size_t bytes) -> void* {
        void* p = ws + off;
        off += (bytes + 255) & ~(size_t)255;
        return p;
    };
    u16*   Xbf  = (u16*)alloc((size_t)NTOK * Dm * 2);
    u16*   W1bf = (u16*)alloc((size_t)DFF * Dm * 2);
    u16*   W3bf = (u16*)alloc((size_t)DFF * Dm * 2);
    u16*   W2bf = (u16*)alloc((size_t)Dm * DFF * 2);
    u16*   Hbf  = (u16*)alloc((size_t)HROW * DFF * 2);
    float* Yadj = (float*)alloc((size_t)HROW * Dm * 4);
    float* T1   = (float*)alloc((size_t)HROW * R * 4);
    float* T3   = (float*)alloc((size_t)HROW * R * 4);
    float* T2   = (float*)alloc((size_t)HROW * R * 4);
    int*   eidx = (int*)alloc((size_t)NMOE * 2 * 4);
    float* eww  = (float*)alloc((size_t)NMOE * 2 * 4);
    (void)ws_size; (void)in_sizes; (void)n_in; (void)out_size;

    // 1) bf16 conversions
    moe_cvt_bf16<<<2048, 256, 0, stream>>>(data, Xbf, (long)NTOK * Dm);
    moe_cvt_bf16<<<4096, 256, 0, stream>>>(W1, W1bf, (long)DFF * Dm);
    moe_cvt_bf16<<<4096, 256, 0, stream>>>(W3, W3bf, (long)DFF * Dm);
    moe_cvt_bf16<<<4096, 256, 0, stream>>>(W2, W2bf, (long)Dm * DFF);

    // 2) gating
    moe_gate<<<NMOE, 256, 0, stream>>>(data, gateW, logits);
    moe_topk<<<(NMOE + 255) / 256, 256, 0, stream>>>(logits, eidx, eww);

    // 3) rank-16 x-side LoRA contractions
    moe_lora_x<<<HROW, 256, 0, stream>>>(data, A1, A3, la1, la3, eidx, T1, T3);

    // 4) fused up/gate tiled WMMA GEMM (TDM-staged) + SiLU -> Hbf
    moe_gemm_gu<<<dim3(DFF / BN, NTOK / BM), 256, 0, stream>>>(
        Xbf, W1bf, W3bf, T1, T3, B1, B3, lb1, lb3, eidx, Hbf);

    // 5) rank-16 h-side LoRA contraction
    moe_t2<<<HROW, 256, 0, stream>>>(Hbf, A2, la2, eidx, T2);

    // 6) down tiled WMMA GEMM (TDM-staged) + LoRA delta
    moe_gemm_y<<<dim3(Dm / BN, HROW / BM), 256, 0, stream>>>(
        Hbf, W2bf, T2, B2, lb2, eidx, Yadj);

    // 7) weighted combine + dense copy
    moe_combine<<<(int)(((long)NTOK * Dm + 255) / 256), 256, 0, stream>>>(Yadj, eww, out);
}
